// L1_EmbbedingGNN_1717986918540
// MI455X (gfx1250) — matrix operand order repaired and checked by reading, not compiled, not run on hardware
//
#include <hip/hip_runtime.h>
#include <hip/hip_bf16.h>

// ---------- vector types matching CDNA5 WMMA operand shapes ----------
typedef __bf16 v16bf __attribute__((ext_vector_type(16)));
typedef __bf16 v8bf  __attribute__((ext_vector_type(8)));
typedef float  v8f   __attribute__((ext_vector_type(8)));

#define WAVES_PER_BLOCK 2

// intra-wave LDS producer->consumer fence (one wave owns its LDS slice;
// DScnt covers all 32 lanes' DS ops, so no workgroup barrier is needed)
__device__ __forceinline__ void lds_fence() {
    __builtin_amdgcn_wave_barrier();
    asm volatile("s_wait_dscnt 0" ::: "memory");
}

// ------------------------------------------------------------------
// zero fill (graph-capture safe; re-run every launch for determinism)
// ------------------------------------------------------------------
__global__ void zero_f32(float* __restrict__ p, long n) {
    long i = (long)blockIdx.x * blockDim.x + threadIdx.x;
    long stride = (long)gridDim.x * blockDim.x;
    for (; i < n; i += stride) p[i] = 0.0f;
}

// ------------------------------------------------------------------
// segment-sum: one 32-lane group per edge, lane = feature dim.
// Rows live in L2 (items 25.6MB / ops 64MB / aggs 51MB all < 192MB L2),
// so the f32 atomics resolve at L2 speed.
// ------------------------------------------------------------------
__global__ void scatter_add32(const float* __restrict__ src,
                              const int* __restrict__ dst_idx,
                              const int* __restrict__ src_idx,
                              float* __restrict__ out, long nedges) {
    long t = (long)blockIdx.x * blockDim.x + threadIdx.x;
    long e = t >> 5;
    int  d = (int)(t & 31);
    if (e >= nedges) return;
    int de = dst_idx[e];
    int se = src_idx[e];
    atomicAdd(out + (size_t)de * 32 + d, src[(size_t)se * 32 + d]);
}

// ------------------------------------------------------------------
// Repack f32 weight matrix W[K][N] into bf16 WMMA B-fragments.
// Tile (kt,nt) is 32(K)x16(N); lane l holds column n=l&15, K =
// 8*(l>>4)+e (e<8), 16+8*(l>>4)+(e-8) (e>=8) -> 16 bf16 (32B) per lane.
// ------------------------------------------------------------------
__global__ void pack_w(const float* __restrict__ w, __bf16* __restrict__ dst,
                       int K, int Nn) {
    int total = K * Nn;
    int p = blockIdx.x * blockDim.x + threadIdx.x;
    if (p >= total) return;
    int ntiles = Nn >> 4;
    int tile   = p >> 9;          // / 512
    int within = p & 511;
    int lane   = within >> 4;
    int e      = within & 15;
    int kt = tile / ntiles;
    int nt = tile - kt * ntiles;
    int hi = lane >> 4;
    int n  = lane & 15;
    int k  = kt * 32 + (e < 8 ? hi * 8 + e : 16 + hi * 8 + (e - 8));
    dst[p] = (__bf16)w[(size_t)k * Nn + nt * 16 + n];
}

// ------------------------------------------------------------------
// A-fragment from an LDS activation buffer (row-major [16][stride] bf16).
// 16-bit A layout: lane l = row m=l&15, khalf=l>>4; elements are two
// contiguous 8-bf16 chunks at K = khalf*8 and K = 16 + khalf*8.
// ------------------------------------------------------------------
__device__ __forceinline__ v16bf load_afrag(const __bf16* in, int in_stride,
                                            int kt, int lane) {
    int m  = lane & 15;
    int hi = lane >> 4;
    const __bf16* base = in + m * in_stride + kt * 32 + hi * 8;
    v8bf lo = *(const v8bf*)base;
    v8bf hh = *(const v8bf*)(base + 16);
    return __builtin_shufflevector(lo, hh, 0,1,2,3,4,5,6,7,8,9,10,11,12,13,14,15);
}

// One dense layer for a 16-row tile: [16,K] @ [K,N] + bias (+optional ELU),
// bf16 in / f32 accumulate via v_wmma_f32_16x16x32_bf16, bf16 out to LDS.
__device__ __forceinline__ void mlp_layer(const __bf16* in, int in_stride, int K,
                                          __bf16* outb, int out_stride, int Nn,
                                          const __bf16* __restrict__ wp,
                                          const float* __restrict__ bias,
                                          int lane, bool act) {
    int ktiles = K >> 5;
    int ntiles = Nn >> 4;
    int hi = lane >> 4, nl = lane & 15;
    for (int nt = 0; nt < ntiles; ++nt) {
        v8f c = {0.f,0.f,0.f,0.f,0.f,0.f,0.f,0.f};
        for (int kt = 0; kt < ktiles; ++kt) {
            v16bf a = load_afrag(in, in_stride, kt, lane);
            v16bf b = *(const v16bf*)(wp + ((size_t)(kt * ntiles + nt) * 32 + lane) * 16);
            c = __builtin_amdgcn_wmma_f32_16x16x32_bf16(false, a, false, b,
                                                        (short)0, c, false, false);
        }
        float bn = bias[nt * 16 + nl];
        #pragma unroll
        for (int r = 0; r < 8; ++r) {
            int m = hi * 8 + r;                       // C layout: lanes16-31 hold M=8..15
            float v = c[r] + bn;
            if (act) v = (v > 0.f) ? v : expm1f(v);   // ELU(alpha=1)
            outb[m * out_stride + nt * 16 + nl] = (__bf16)v;
        }
    }
}

// Gather/load a 16x32 f32 tile into LDS as bf16 (lane -> half row).
__device__ __forceinline__ void load_tile32(const float* __restrict__ src,
                                            const int* __restrict__ gather,
                                            long rowbase, __bf16* xin, int lane) {
    int row = lane >> 1;
    int cb  = (lane & 1) * 16;
    long grow = rowbase + row;
    long srow = gather ? (long)gather[grow] : grow;
    const float4* p4 = (const float4*)(src + srow * 32 + cb);
    float4 f0 = p4[0], f1 = p4[1], f2 = p4[2], f3 = p4[3];
    v16bf v;
    v[0]=(__bf16)f0.x; v[1]=(__bf16)f0.y; v[2]=(__bf16)f0.z; v[3]=(__bf16)f0.w;
    v[4]=(__bf16)f1.x; v[5]=(__bf16)f1.y; v[6]=(__bf16)f1.z; v[7]=(__bf16)f1.w;
    v[8]=(__bf16)f2.x; v[9]=(__bf16)f2.y; v[10]=(__bf16)f2.z; v[11]=(__bf16)f2.w;
    v[12]=(__bf16)f3.x; v[13]=(__bf16)f3.y; v[14]=(__bf16)f3.z; v[15]=(__bf16)f3.w;
    *(v16bf*)(xin + row * 32 + cb) = v;
}

struct FusedArgs {
    const float* items;
    const int*   parents;
    const float* aggc;
    const float* aggo;
    const __bf16* wpack;
    const float* bias[15];   // [mlp*3 + layer]; mlp: 0=self 1=parent 2=children 3=ops 4=combined
    float* out;
    int n;
};

// ------------------------------------------------------------------
// Fully fused GNN: one wave = one 16-row tile through all 5 MLPs.
// LDS per wave: xin 16x32 + t0/t1 16x128 + xc 16x256 (bf16) = 17KB.
// Weight fragments stream from global (344KB total, WGP$/L2 resident).
// ------------------------------------------------------------------
__global__ void __launch_bounds__(WAVES_PER_BLOCK * 32)
fused_gnn(FusedArgs A) {
    __shared__ __align__(32) __bf16 smem[WAVES_PER_BLOCK * (512 + 2048 + 2048 + 4096)];
    int lane = threadIdx.x & 31;
    int wid  = threadIdx.x >> 5;
    __bf16* base = smem + wid * (512 + 2048 + 2048 + 4096);
    __bf16* xin = base;                 // 16x32
    __bf16* t0  = base + 512;           // 16x128
    __bf16* t1  = base + 512 + 2048;    // 16x128
    __bf16* xc  = base + 512 + 4096;    // 16x256 (concat: parent|children|ops|self)

    long tile = (long)blockIdx.x * WAVES_PER_BLOCK + wid;
    long rowbase = tile * 16;
    if (rowbase >= A.n) return;

    const long wbase[5] = {0, 28672, 57344, 86016, 114688};

    // concat order in reference: [parent, children, ops, self]
    for (int br = 0; br < 4; ++br) {
        const float* src; const int* gather = nullptr; int mlp;
        if (br == 0)      { src = A.items; gather = A.parents; mlp = 1; }
        else if (br == 1) { src = A.aggc;  mlp = 2; }
        else if (br == 2) { src = A.aggo;  mlp = 3; }
        else              { src = A.items; mlp = 0; }

        load_tile32(src, gather, rowbase, xin, lane);
        lds_fence();
        const __bf16* wp = A.wpack + wbase[mlp];
        mlp_layer(xin, 32, 32,  t0, 128, 128, wp,                 A.bias[mlp*3+0], lane, true);
        lds_fence();
        mlp_layer(t0, 128, 128, t1, 128, 128, wp + 4096,          A.bias[mlp*3+1], lane, true);
        lds_fence();
        mlp_layer(t1, 128, 128, xc + br * 64, 256, 64, wp + 20480, A.bias[mlp*3+2], lane, false);
        lds_fence();
    }

    // combined MLP on [16,256]
    const __bf16* cw = A.wpack + wbase[4];
    mlp_layer(xc, 256, 256, t0, 128, 128, cw,          A.bias[12], lane, true);
    lds_fence();
    mlp_layer(t0, 128, 128, t1, 128, 128, cw + 32768,  A.bias[13], lane, true);
    lds_fence();

    // final layer: [16,128]@[128,64]+b -> f32 global; row n-1 forced to zero
    {
        const __bf16* wp   = cw + 32768 + 16384;
        const float*  bias = A.bias[14];
        int hi = lane >> 4, nl = lane & 15;
        for (int nt = 0; nt < 4; ++nt) {
            v8f c = {0.f,0.f,0.f,0.f,0.f,0.f,0.f,0.f};
            for (int kt = 0; kt < 4; ++kt) {
                v16bf a = load_afrag(t1, 128, kt, lane);
                v16bf b = *(const v16bf*)(wp + ((size_t)(kt * 4 + nt) * 32 + lane) * 16);
                c = __builtin_amdgcn_wmma_f32_16x16x32_bf16(false, a, false, b,
                                                            (short)0, c, false, false);
            }
            float bn = bias[nt * 16 + nl];
            #pragma unroll
            for (int r = 0; r < 8; ++r) {
                int m = hi * 8 + r;
                long grow = rowbase + m;
                float v = (grow == (long)A.n - 1) ? 0.f : (c[r] + bn);
                A.out[grow * 64 + nt * 16 + nl] = v;
            }
        }
    }
}

// ------------------------------------------------------------------
extern "C" void kernel_launch(void* const* d_in, const int* in_sizes, int n_in,
                              void* d_out, int out_size, void* d_ws, size_t ws_size,
                              hipStream_t stream) {
    const float* items = (const float*)d_in[0];
    const int*   parents = (const int*)d_in[1];
    const float* ops   = (const float*)d_in[2];
    const int*   iedge = (const int*)d_in[3];   // [2,E1] row0=dst(item) row1=src(child)
    const int*   oedge = (const int*)d_in[4];   // [2,E2] row0=dst(item) row1=src(op)

    int  N  = in_sizes[0] / 32;
    long E1 = (long)in_sizes[3] / 2;
    long E2 = (long)in_sizes[4] / 2;

    // workspace layout: agg_children [N,32] f32 | agg_ops [N,32] f32 | packed bf16 weights
    float*  aggc  = (float*)d_ws;
    float*  aggo  = aggc + (size_t)N * 32;
    __bf16* wpack = (__bf16*)(aggo + (size_t)N * 32);

    zero_f32<<<4096, 256, 0, stream>>>(aggc, (long)N * 64);

    {
        long threads = E1 * 32;
        scatter_add32<<<(int)((threads + 255) / 256), 256, 0, stream>>>(
            items, iedge, iedge + E1, aggc, E1);
    }
    {
        long threads = E2 * 32;
        scatter_add32<<<(int)((threads + 255) / 256), 256, 0, stream>>>(
            ops, oedge, oedge + E2, aggo, E2);
    }

    // pack all 15 weight matrices into WMMA B-fragment layout
    const long wbase[5] = {0, 28672, 57344, 86016, 114688};
    const int  Ks[5][3] = {{32,128,128},{32,128,128},{32,128,128},{32,128,128},{256,128,128}};
    const int  Ns[5][3] = {{128,128,64},{128,128,64},{128,128,64},{128,128,64},{128,128,64}};
    for (int m = 0; m < 5; ++m) {
        long o = wbase[m];
        for (int l = 0; l < 3; ++l) {
            const float* w = (const float*)d_in[5 + m * 6 + l * 2];
            int K = Ks[m][l], Nn = Ns[m][l];
            int total = K * Nn;
            pack_w<<<(total + 255) / 256, 256, 0, stream>>>(w, wpack + o, K, Nn);
            o += (long)K * Nn;
        }
    }

    FusedArgs A;
    A.items = items; A.parents = parents; A.aggc = aggc; A.aggo = aggo;
    A.wpack = wpack; A.out = (float*)d_out; A.n = N;
    for (int m = 0; m < 5; ++m)
        for (int l = 0; l < 3; ++l)
            A.bias[m * 3 + l] = (const float*)d_in[5 + m * 6 + l * 2 + 1];

    int tiles  = (N + 15) / 16;
    int blocks = (tiles + WAVES_PER_BLOCK - 1) / WAVES_PER_BLOCK;
    fused_gnn<<<blocks, WAVES_PER_BLOCK * 32, 0, stream>>>(A);
}
